// MambaIntegerBlockV2Full_62191126446735
// MI455X (gfx1250) — matrix operand order repaired and checked by reading, not compiled, not run on hardware
//
#include <hip/hip_runtime.h>

// Problem constants (from reference): B=4, H=16 -> BH=64; S=4096; dh=64; ds=128; chunk=128
#define BH    64
#define S_LEN 4096
#define DH    64
#define DS    128
#define CS    128
#define NC    32

#define Y_ELEMS   ((size_t)BH * S_LEN * DH)   // 16,777,216 f32
#define H_ELEMS   ((size_t)BH * DS * DH)      // 524,288 f32
#define WS_CONTRIB_ELEMS ((size_t)BH * NC * DS * DH) // 16,777,216 f32 (64 MB)

typedef __bf16 v16bf __attribute__((ext_vector_type(16)));
typedef float  v8f   __attribute__((ext_vector_type(8)));
typedef unsigned int u32x4 __attribute__((ext_vector_type(4)));
typedef unsigned int u32x8 __attribute__((ext_vector_type(8)));

struct Q2 { uint4 a, b; };

__device__ __forceinline__ unsigned short f2bf(float f) {
  unsigned u = __builtin_bit_cast(unsigned, f);
  return (unsigned short)((u + 0x7FFFu + ((u >> 16) & 1u)) >> 16); // RNE truncate to bf16
}

// 16-bit A fragment (16x32, MxK): lane half (lane>=16) offsets K by +8;
// elements 0..7 hold K = kbase+8*half .. +7, elements 8..15 hold K = kbase+16+8*half .. +7
__device__ __forceinline__ v16bf frag_a(const unsigned short* rowPtr, int kbase, int half) {
  const unsigned short* p = rowPtr + kbase + 8 * half;
  Q2 q{ *(const uint4*)(p), *(const uint4*)(p + 16) };
  return __builtin_bit_cast(v16bf, q);
}
// 16-bit B fragment (32x16, KxN): same K order as A: elements 0..7 K=kbase+8*half..+7, 8..15 K=kbase+16+8*half..+7
__device__ __forceinline__ v16bf frag_b(const unsigned short* colPtr, int kbase, int half) {
  const unsigned short* p = colPtr + kbase + 8 * half;
  Q2 q{ *(const uint4*)(p), *(const uint4*)(p + 16) };
  return __builtin_bit_cast(v16bf, q);
}

__device__ __forceinline__ v8f wmma_bf16(v16bf a, v16bf b, v8f c) {
  return __builtin_amdgcn_wmma_f32_16x16x32_bf16(false, a, false, b, (short)0, c, false, false);
}

// ---------------------------------------------------------------------------
// Tensor Data Mover: DMA `nelem` f32 (1 row) from global into LDS.
// Issued by the calling wave only (EXEC-independent); tracked by TENSORcnt.
// D# built in wave-uniform registers, bound as SGPR tuples.
// ---------------------------------------------------------------------------
__device__ __forceinline__ void tdm_load_row_f32(const float* gsrc, float* ldsDst,
                                                 unsigned nelem) {
  unsigned long long ga = (unsigned long long)gsrc;
  unsigned ldsAddr = (unsigned)(unsigned long long)(uintptr_t)ldsDst; // low 32 bits = LDS offset

  // Group 0: count=1 | lds_addr | global_addr[56:0] | type=2 ("image")
  u32x4 g0;
  g0.x = 1u;                                          // count=1, user desc
  g0.y = ldsAddr;                                     // lds_addr
  g0.z = (unsigned)(ga & 0xFFFFFFFFull);              // global_addr[31:0]
  g0.w = (unsigned)((ga >> 32) & 0x01FFFFFFull) | (2u << 30); // global_addr[56:32] | type=2

  // Group 1: data_size=4B (code 2), tile = nelem x 1, tensor dims = nelem x 1
  unsigned long long q0 = (2ull << 16)                          // data_size=2 (4 bytes)
                        | ((unsigned long long)(nelem & 0xFFFFu) << 48); // tensor_dim0[15:0]
  unsigned long long q1 = ((unsigned long long)(nelem >> 16) & 0xFFFFull)   // tensor_dim0[31:16]
                        | (1ull << 16)                                      // tensor_dim1 = 1
                        | ((unsigned long long)(nelem & 0xFFFFu) << 48);    // tile_dim0
  unsigned long long q2 = 1ull                                   // tile_dim1 = 1
                        | ((unsigned long long)nelem << 32);     // tensor_dim0_stride[31:0]
  unsigned long long q3 = 0ull;                                  // stride hi bits / dim1 stride = 0
  u32x8 g1;
  g1.s0 = (unsigned)q0; g1.s1 = (unsigned)(q0 >> 32);
  g1.s2 = (unsigned)q1; g1.s3 = (unsigned)(q1 >> 32);
  g1.s4 = (unsigned)q2; g1.s5 = (unsigned)(q2 >> 32);
  g1.s6 = (unsigned)q3; g1.s7 = (unsigned)(q3 >> 32);

  asm volatile("tensor_load_to_lds %0, %1" :: "s"(g0), "s"(g1) : "memory");
  __builtin_amdgcn_s_wait_tensorcnt(0);
}

// Inclusive cumsum of 128 values already resident in cum[] via Hillis-Steele in LDS.
__device__ __forceinline__ void lds_cumsum128(float* cum, int tid) {
  for (int off = 1; off < CS; off <<= 1) {
    float v = 0.f;
    if (tid < CS) { v = cum[tid]; if (tid >= off) v += cum[tid - off]; }
    __syncthreads();
    if (tid < CS) cum[tid] = v;
    __syncthreads();
  }
}

// ---------------------------------------------------------------------------
// Pass 1: per-(bh,chunk) intra-chunk work.
//   W   = L ∘ (C·B^T)           (128x128, bf16 in LDS)
//   Y   = W · X                 (written to d_out Y region)
//   Hc  = (B*e_rev)^T · X       (written to ws contrib slot)
//   decay_total -> ws
// LDS layout (ushort): Cc[16384] Bc[16384] BcT[16384] W[16384] XcT[8192]
//                     + cum[128]f32 + erev[128]f32  => 148480 bytes
// ---------------------------------------------------------------------------
__global__ __launch_bounds__(256, 1)
void ssd_pass1(const float* __restrict__ Xg, const float* __restrict__ Ag,
               const float* __restrict__ Bg, const float* __restrict__ Cg,
               float* __restrict__ Yout, float* __restrict__ wsContrib,
               float* __restrict__ wsDecay) {
  const int tid  = threadIdx.x;
  const int bh   = blockIdx.x >> 5;
  const int c    = blockIdx.x & 31;
  const size_t rowBase = (size_t)bh * S_LEN + (size_t)c * CS; // chunk's first row index

  extern __shared__ __align__(16) unsigned char smem[];
  unsigned short* Cc  = (unsigned short*)smem;    // [i][d]  128x128
  unsigned short* Bc  = Cc  + 16384;              // [t][d]  128x128
  unsigned short* BcT = Bc  + 16384;              // [d][t]  pre-scaled by e_rev[t]
  unsigned short* W   = BcT + 16384;              // [i][j]  L ∘ (C B^T)
  unsigned short* XcT = W   + 16384;              // [d][t]  64x128
  float* cum  = (float*)(XcT + 8192);             // 128
  float* erev = cum + 128;                        // 128

  const float* Cgc = Cg + rowBase * DS;
  const float* Bgc = Bg + rowBase * DS;
  const float* Xgc = Xg + rowBase * DH;

  __builtin_prefetch(Cgc, 0, 0);
  __builtin_prefetch(Bgc, 0, 0);
  __builtin_prefetch(Xgc, 0, 0);

  // ---- DMA the chunk's A vector into LDS via the Tensor Data Mover ----
  if (tid < 32) {                 // wave 0 issues one TDM op; TENSORcnt tracks it
    tdm_load_row_f32(Ag + rowBase, cum, CS);
  }
  __syncthreads();
  lds_cumsum128(cum, tid);
  if (tid < CS) erev[tid] = __expf(cum[CS - 1] - cum[tid]);
  __syncthreads();

  // ---- stage operands in LDS as bf16 ----
  // C chunk: linear copy with conversion
#pragma unroll
  for (int r = 0; r < 16; ++r) {
    int e4 = tid + 256 * r;            // float4 index, 0..4095
    int base = e4 * 4;                 // element index == i*128+d
    float4 v = ((const float4*)Cgc)[e4];
    unsigned h0 = f2bf(v.x), h1 = f2bf(v.y), h2 = f2bf(v.z), h3 = f2bf(v.w);
    *(uint2*)(Cc + base) = make_uint2(h0 | (h1 << 16), h2 | (h3 << 16));
  }
  // B chunk: natural + transposed (scaled by e_rev[t])
#pragma unroll
  for (int r = 0; r < 16; ++r) {
    int e4 = tid + 256 * r;
    int base = e4 * 4;
    int t = base >> 7, d = base & 127;
    float4 v = ((const float4*)Bgc)[e4];
    unsigned h0 = f2bf(v.x), h1 = f2bf(v.y), h2 = f2bf(v.z), h3 = f2bf(v.w);
    *(uint2*)(Bc + base) = make_uint2(h0 | (h1 << 16), h2 | (h3 << 16));
    float er = erev[t];
    BcT[(d + 0) * 128 + t] = f2bf(v.x * er);
    BcT[(d + 1) * 128 + t] = f2bf(v.y * er);
    BcT[(d + 2) * 128 + t] = f2bf(v.z * er);
    BcT[(d + 3) * 128 + t] = f2bf(v.w * er);
  }
  // X chunk: transposed [d][t]
#pragma unroll
  for (int r = 0; r < 8; ++r) {
    int e4 = tid + 256 * r;
    int base = e4 * 4;
    int t = base >> 6, d = base & 63;
    float4 v = ((const float4*)Xgc)[e4];
    XcT[(d + 0) * 128 + t] = f2bf(v.x);
    XcT[(d + 1) * 128 + t] = f2bf(v.y);
    XcT[(d + 2) * 128 + t] = f2bf(v.z);
    XcT[(d + 3) * 128 + t] = f2bf(v.w);
  }
  // zero W (strictly-upper tiles are never written by GEMM1)
#pragma unroll
  for (int r = 0; r < 8; ++r)
    ((uint4*)W)[tid + 256 * r] = make_uint4(0u, 0u, 0u, 0u);
  __syncthreads();

  const int wid  = tid >> 5;
  const int lane = tid & 31;
  const int ln   = lane & 15;
  const int half = lane >> 4;

  // ---- GEMM1: S1^T tile (M=j, N=i) = B(jt) x C(it)^T ; only it >= jt tiles ----
  for (int p = wid; p < 36; p += 8) {
    int it = 0, accn = 0;
    while (accn + it + 1 <= p) { accn += it + 1; ++it; }
    int jt = p - accn;
    const unsigned short* aRow = Bc + (jt * 16 + ln) * 128; // A: row j, K=d
    const unsigned short* bCol = Cc + (it * 16 + ln) * 128; // B: col i, K=d
    v8f acc = {0.f, 0.f, 0.f, 0.f, 0.f, 0.f, 0.f, 0.f};
#pragma unroll
    for (int kk = 0; kk < 4; ++kk)
      acc = wmma_bf16(frag_a(aRow, kk * 32, half), frag_b(bCol, kk * 32, half), acc);
    // store as W[i][j] with causal decay L[i][j] = exp(cum_i - cum_j), i>=j
    // branchless: vector-load cum[jbase..jbase+7], unconditional expf, cndmask select
    int i = it * 16 + ln;
    int jbase = jt * 16 + 8 * half;
    float ci = cum[i];
    float4 cj0 = *(const float4*)(cum + jbase);
    float4 cj1 = *(const float4*)(cum + jbase + 4);
    float cj[8] = {cj0.x, cj0.y, cj0.z, cj0.w, cj1.x, cj1.y, cj1.z, cj1.w};
    unsigned hh[8];
#pragma unroll
    for (int r = 0; r < 8; ++r) {
      float e = __expf(ci - cj[r]);
      float v = acc[r] * e;
      v = (i >= jbase + r) ? v : 0.f;
      hh[r] = f2bf(v);
    }
    *(uint4*)(W + i * 128 + jbase) =
        make_uint4(hh[0] | (hh[1] << 16), hh[2] | (hh[3] << 16),
                   hh[4] | (hh[5] << 16), hh[6] | (hh[7] << 16));
  }

  // ---- GEMM3: contrib[s][d] = sum_t BcT[s][t] * X[t][d]  (BcT pre-scaled) ----
  float* contrib = wsContrib + ((size_t)(bh * NC + c)) * (DS * DH);
  for (int tl = wid; tl < 32; tl += 8) {
    int st = tl >> 2, dt = tl & 3;
    const unsigned short* aRow = BcT + (st * 16 + ln) * 128;
    const unsigned short* bCol = XcT + (dt * 16 + ln) * 128;
    v8f acc = {0.f, 0.f, 0.f, 0.f, 0.f, 0.f, 0.f, 0.f};
#pragma unroll
    for (int kk = 0; kk < 4; ++kk)
      acc = wmma_bf16(frag_a(aRow, kk * 32, half), frag_b(bCol, kk * 32, half), acc);
    int d = dt * 16 + ln;
#pragma unroll
    for (int r = 0; r < 8; ++r) {
      int s = st * 16 + r + 8 * half;
      contrib[s * DH + d] = acc[r];
    }
  }
  __syncthreads(); // W fully written

  // ---- GEMM2: Y_intra[i][d] = sum_j W[i][j] * X[j][d] ----
  float* Yc = Yout + rowBase * DH;
  for (int tl = wid; tl < 32; tl += 8) {
    int it = tl >> 2, dt = tl & 3;
    const unsigned short* aRow = W + (it * 16 + ln) * 128;
    const unsigned short* bCol = XcT + (dt * 16 + ln) * 128;
    v8f acc = {0.f, 0.f, 0.f, 0.f, 0.f, 0.f, 0.f, 0.f};
#pragma unroll
    for (int kk = 0; kk < 4; ++kk)
      acc = wmma_bf16(frag_a(aRow, kk * 32, half), frag_b(bCol, kk * 32, half), acc);
    int d = dt * 16 + ln;
#pragma unroll
    for (int r = 0; r < 8; ++r) {
      int i = it * 16 + r + 8 * half;
      Yc[(size_t)i * DH + d] = acc[r];
    }
  }

  if (tid == 0) wsDecay[bh * NC + c] = __expf(cum[CS - 1]);
}

// ---------------------------------------------------------------------------
// Pass 2: sequential inter-chunk state scan per (bh). State (128x64 f32) lives
// in registers (32 per thread). Overwrites each contrib slot with the state
// *entering* that chunk; writes final state to d_out's h region.
// ---------------------------------------------------------------------------
__global__ __launch_bounds__(256, 1)
void ssd_pass2(const float* __restrict__ h0, const float* __restrict__ wsDecay,
               float* __restrict__ wsContrib, float* __restrict__ Hout) {
  const int bh = blockIdx.x;
  const int tid = threadIdx.x;
  float h[32];
#pragma unroll
  for (int r = 0; r < 32; ++r)
    h[r] = h0[(size_t)bh * (DS * DH) + tid + 256 * r];
  for (int c = 0; c < NC; ++c) {
    float dt = wsDecay[bh * NC + c];
    float* slot = wsContrib + ((size_t)(bh * NC + c)) * (DS * DH);
#pragma unroll
    for (int r = 0; r < 32; ++r) {
      int f = tid + 256 * r;
      float contrib = slot[f];
      slot[f] = h[r];                 // state entering chunk c (for pass 3)
      h[r] = dt * h[r] + contrib;     // h_end = decay_total*h_prev + contrib
    }
  }
#pragma unroll
  for (int r = 0; r < 32; ++r)
    Hout[(size_t)bh * (DS * DH) + tid + 256 * r] = h[r];
}

// ---------------------------------------------------------------------------
// Pass 3: Y += diag(exp(cum)) * C * h_prev  per (bh,chunk).
// LDS: Csc[16384]us (scale folded in) + hT[8192]us + cum[128]f + efwd[128]f
// ---------------------------------------------------------------------------
__global__ __launch_bounds__(256, 1)
void ssd_pass3(const float* __restrict__ Ag, const float* __restrict__ Cg,
               const float* __restrict__ wsContrib, float* __restrict__ Yout) {
  const int tid = threadIdx.x;
  const int bh  = blockIdx.x >> 5;
  const int c   = blockIdx.x & 31;
  const size_t rowBase = (size_t)bh * S_LEN + (size_t)c * CS;

  extern __shared__ __align__(16) unsigned char smem[];
  unsigned short* Csc = (unsigned short*)smem;   // [i][d], pre-scaled by exp(cum_i)
  unsigned short* hT  = Csc + 16384;             // [d][s]  64x128
  float* cum  = (float*)(hT + 8192);             // 128
  float* efwd = cum + 128;                       // 128

  const float* Cgc = Cg + rowBase * DS;
  const float* hs  = wsContrib + ((size_t)(bh * NC + c)) * (DS * DH);
  __builtin_prefetch(Cgc, 0, 0);
  __builtin_prefetch(hs, 0, 0);

  if (tid < 32) {                 // wave 0: TDM the A vector into LDS
    tdm_load_row_f32(Ag + rowBase, cum, CS);
  }
  __syncthreads();
  lds_cumsum128(cum, tid);
  if (tid < CS) efwd[tid] = __expf(cum[tid]);
  __syncthreads();

#pragma unroll
  for (int r = 0; r < 16; ++r) {
    int e4 = tid + 256 * r;
    int base = e4 * 4;
    int i = base >> 7;
    float ef = efwd[i];
    float4 v = ((const float4*)Cgc)[e4];
    unsigned h0 = f2bf(v.x * ef), h1 = f2bf(v.y * ef);
    unsigned h2 = f2bf(v.z * ef), h3 = f2bf(v.w * ef);
    *(uint2*)(Csc + base) = make_uint2(h0 | (h1 << 16), h2 | (h3 << 16));
  }
#pragma unroll
  for (int r = 0; r < 8; ++r) {
    int e4 = tid + 256 * r;
    int base = e4 * 4;
    int s = base >> 6, d = base & 63;
    float4 v = ((const float4*)hs)[e4];
    hT[(d + 0) * 128 + s] = f2bf(v.x);
    hT[(d + 1) * 128 + s] = f2bf(v.y);
    hT[(d + 2) * 128 + s] = f2bf(v.z);
    hT[(d + 3) * 128 + s] = f2bf(v.w);
  }
  __syncthreads();

  const int wid  = tid >> 5;
  const int lane = tid & 31;
  const int ln   = lane & 15;
  const int half = lane >> 4;

  float* Yc = Yout + rowBase * DH;
  for (int tl = wid; tl < 32; tl += 8) {
    int it = tl >> 2, dt = tl & 3;
    const unsigned short* aRow = Csc + (it * 16 + ln) * 128;
    const unsigned short* bCol = hT + (dt * 16 + ln) * 128;
    v8f acc = {0.f, 0.f, 0.f, 0.f, 0.f, 0.f, 0.f, 0.f};
#pragma unroll
    for (int kk = 0; kk < 4; ++kk)
      acc = wmma_bf16(frag_a(aRow, kk * 32, half), frag_b(bCol, kk * 32, half), acc);
    int d = dt * 16 + ln;
#pragma unroll
    for (int r = 0; r < 8; ++r) {
      int i = it * 16 + r + 8 * half;
      size_t yi = (size_t)i * DH + d;
      Yc[yi] = Yc[yi] + acc[r];
    }
  }
}

extern "C" void kernel_launch(void* const* d_in, const int* in_sizes, int n_in,
                              void* d_out, int out_size, void* d_ws, size_t ws_size,
                              hipStream_t stream) {
  const float* X  = (const float*)d_in[0];
  const float* A  = (const float*)d_in[1];
  const float* Bm = (const float*)d_in[2];
  const float* C  = (const float*)d_in[3];
  const float* h0 = (const float*)d_in[4];

  float* Y    = (float*)d_out;
  float* Hout = Y + Y_ELEMS;

  float* wsContrib = (float*)d_ws;                 // 64 MB: per-chunk contrib, then per-chunk entering state
  float* wsDecay   = wsContrib + WS_CONTRIB_ELEMS; // 2048 f32: per-chunk total decay

  const size_t sm1 = (size_t)(16384 * 4 + 8192) * 2 + 256 * 4; // 148480 B
  const size_t sm3 = (size_t)(16384 + 8192) * 2 + 256 * 4;     //  50176 B

  ssd_pass1<<<dim3(BH * NC), dim3(256), sm1, stream>>>(X, A, Bm, C, Y, wsContrib, wsDecay);
  ssd_pass2<<<dim3(BH), dim3(256), 0, stream>>>(h0, wsDecay, wsContrib, Hout);
  ssd_pass3<<<dim3(BH * NC), dim3(256), sm3, stream>>>(A, C, wsContrib, Y);
}
